// MultiHeadAttention_60610578481681
// MI455X (gfx1250) — matrix operand order repaired and checked by reading, not compiled, not run on hardware
//
#include <hip/hip_runtime.h>
#include <hip/hip_bf16.h>

typedef __attribute__((ext_vector_type(16))) __bf16        v16bf;
typedef __attribute__((ext_vector_type(8)))  float         v8f;
typedef __attribute__((ext_vector_type(4)))  unsigned int  v4u;
typedef __attribute__((ext_vector_type(8)))  int           v8i_;
typedef __attribute__((ext_vector_type(4)))  int           v4i_;

#define B_  2
#define T_  2048
#define D_  1024
#define H_  16
#define HD_ 64
#define BT_ (B_ * T_)

// GEMM macro-tile config
#define BM  128
#define BN  128
#define BK  32
#define LDT 40            // LDS row stride in bf16 units (32 payload + 8 pad -> 80B, 16B aligned)

// ---------------------------------------------------------------------------
// Fragment loaders (bf16 memory / LDS -> WMMA fragments), per CDNA5 layouts.
// A 16x32: lane L holds row M=L&15; element e -> K = (e&8)*2 + (L>>4)*8 + (e&7)
//          => two contiguous 16B chunks at +0 and +16 elements.
// B 32x16 from B^T storage (row n holds K contiguously): lane L col N=L&15,
//          element e -> K = (e&8)*2 + (L>>4)*8 + (e&7) (same as A).
// ---------------------------------------------------------------------------
static __device__ __forceinline__ v16bf load_A_frag(const __bf16* __restrict__ base, int ld) {
    const int lane = threadIdx.x & 31;
    const __bf16* p = base + (size_t)(lane & 15) * ld + (lane >> 4) * 8;
    union { v16bf v; uint4 q[2]; } u;
    u.q[0] = *(const uint4*)(p);
    u.q[1] = *(const uint4*)(p + 16);
    return u.v;
}

static __device__ __forceinline__ v16bf load_Bt_frag(const __bf16* __restrict__ base, int ld) {
    const int lane = threadIdx.x & 31;
    const __bf16* p = base + (size_t)(lane & 15) * ld + (lane >> 4) * 8;
    union { v16bf v; uint4 q[2]; } u;
    u.q[0] = *(const uint4*)(p);
    u.q[1] = *(const uint4*)(p + 16);
    return u.v;
}

static __device__ __forceinline__ v8f wmma_bf16(v16bf a, v16bf b, v8f c) {
    return __builtin_amdgcn_wmma_f32_16x16x32_bf16(false, a, false, b, (short)0, c, false, false);
}

// ---------------------------------------------------------------------------
// TDM: 2-D bf16 tile (tile_d1 rows x tile_d0 elems) global -> LDS, with
// 4-DWORD pad after every 16 payload DWORDs (row stride LDT bf16 = 80B).
// D# packing per CDNA5 ISA. 6-arg builtin form.
// ---------------------------------------------------------------------------
static __device__ __forceinline__ void tdm_load_tile_bf16(
    const __bf16* gbase, unsigned lds_off,
    unsigned tensor_d0, unsigned tensor_d1, unsigned stride_d0,
    unsigned tile_d0, unsigned tile_d1)
{
    const unsigned long long ga = (unsigned long long)(uintptr_t)gbase;
    v4u g0;
    g0[0] = 1u;                                            // count=1 (valid), user mode
    g0[1] = lds_off;                                       // LDS byte address
    g0[2] = (unsigned)(ga & 0xffffffffu);                  // global addr [31:0]
    g0[3] = (unsigned)((ga >> 32) & 0x01ffffffu) | (2u << 30); // [56:32] + type=2
    v8i_ g1;
    // data_size=1 (2B) | pad_enable | pad_interval=3 (16 DW) | pad_amount=3 (4 DW)
    g1[0] = (int)((1u << 16) | (1u << 20) | (3u << 22) | (3u << 25));
    g1[1] = (int)((tensor_d0 & 0xffffu) << 16);            // tensor_dim0[15:0] @ bits63:48
    g1[2] = (int)(((tensor_d0 >> 16) & 0xffffu) | ((tensor_d1 & 0xffffu) << 16));
    g1[3] = (int)(((tensor_d1 >> 16) & 0xffffu) | ((tile_d0 & 0xffffu) << 16));
    g1[4] = (int)(tile_d1 & 0xffffu);                      // tile_dim1 (tile_dim2=0)
    g1[5] = (int)stride_d0;                                // tensor_dim0_stride[31:0]
    g1[6] = 0;
    g1[7] = 0;
    const v4i_ z4 = {0, 0, 0, 0};
    const v8i_ z8 = {0, 0, 0, 0, 0, 0, 0, 0};
    __builtin_amdgcn_tensor_load_to_lds(g0, g1, z4, z4, z8, 0);
}

// ---------------------------------------------------------------------------
// Shared GEMM mainloop: C[128x128] += A[128xK] * B^T[128xK]^T
// 8 waves in 2(M) x 4(N); per wave: 4 Mx16 x 2 Nx16 accumulators.
// TDM double-buffered LDS staging, s_wait_tensorcnt pipeline.
// ---------------------------------------------------------------------------
static __device__ __forceinline__ void gemm_mainloop(
    const __bf16* __restrict__ A, const __bf16* __restrict__ Bt,
    int row0, int col0, int Kdim, int lda, int ldb,
    __bf16* As, __bf16* Bs,              // each: [2][BM*LDT]
    v8f acc[4][2])
{
    const int wave = threadIdx.x >> 5;
    const int wm   = wave >> 2;          // 0..1
    const int wn   = wave & 3;           // 0..3
    const bool issuer = (wave == 0);
    const int nk = Kdim / BK;

#pragma unroll
    for (int mt = 0; mt < 4; ++mt)
#pragma unroll
        for (int nt = 0; nt < 2; ++nt) acc[mt][nt] = (v8f){};

    const unsigned aoff0 = (unsigned)(uintptr_t)(void*)As;
    const unsigned boff0 = (unsigned)(uintptr_t)(void*)Bs;
    const unsigned bufstride = BM * LDT * 2u;   // bytes per buffer

    if (issuer) {
        tdm_load_tile_bf16(A  + (size_t)row0 * lda, aoff0, lda, BT_, lda, BK, BM);
        tdm_load_tile_bf16(Bt + (size_t)col0 * ldb, boff0, ldb, D_,  ldb, BK, BM);
    }

    for (int kt = 0; kt < nk; ++kt) {
        const int cur = kt & 1;
        if (issuer) {
            if (kt + 1 < nk) {
                const int nxt = (kt + 1) & 1;
                const int k1 = (kt + 1) * BK;
                tdm_load_tile_bf16(A  + (size_t)row0 * lda + k1, aoff0 + nxt * bufstride,
                                   lda, BT_, lda, BK, BM);
                tdm_load_tile_bf16(Bt + (size_t)col0 * ldb + k1, boff0 + nxt * bufstride,
                                   ldb, D_, ldb, BK, BM);
                __builtin_amdgcn_s_wait_tensorcnt((short)2);
            } else {
                __builtin_amdgcn_s_wait_tensorcnt((short)0);
            }
        }
        __syncthreads();                 // publish LDS tile to all waves

        const __bf16* At = As + (size_t)cur * BM * LDT;
        const __bf16* Bl = Bs + (size_t)cur * BM * LDT;

        v16bf af[4], bfrag[2];
#pragma unroll
        for (int mt = 0; mt < 4; ++mt)
            af[mt] = load_A_frag(At + (size_t)(wm * 64 + mt * 16) * LDT, LDT);
#pragma unroll
        for (int nt = 0; nt < 2; ++nt)
            bfrag[nt] = load_Bt_frag(Bl + (size_t)(wn * 32 + nt * 16) * LDT, LDT);

#pragma unroll
        for (int mt = 0; mt < 4; ++mt)
#pragma unroll
            for (int nt = 0; nt < 2; ++nt)
                acc[mt][nt] = wmma_bf16(af[mt], bfrag[nt], acc[mt][nt]);

        __syncthreads();                 // all reads done before buffer refill
    }
}

// ---------------------------------------------------------------------------
// Kernel 0a: cast x (fp32) -> bf16, straight layout [BT, D]
// ---------------------------------------------------------------------------
__global__ __launch_bounds__(256)
void cast_x_kernel(const float* __restrict__ in, __bf16* __restrict__ out) {
    const size_t i0 = ((size_t)blockIdx.x * 256 + threadIdx.x) * 8;
    union { __bf16 h[8]; uint4 q; } u;
#pragma unroll
    for (int e = 0; e < 8; ++e) u.h[e] = (__bf16)in[i0 + e];
    *(uint4*)(out + i0) = u.q;
}

// ---------------------------------------------------------------------------
// Kernel 0b: cast + transpose weights: Wt[n][k] = (bf16)W[k][n], 32x32 tiles
// ---------------------------------------------------------------------------
__global__ __launch_bounds__(256)
void transpose_cast_kernel(const float* __restrict__ W0, const float* __restrict__ W1,
                           const float* __restrict__ W2, const float* __restrict__ W3,
                           __bf16* __restrict__ O0, __bf16* __restrict__ O1,
                           __bf16* __restrict__ O2, __bf16* __restrict__ O3) {
    __shared__ __bf16 tile[32][33];
    const float* W = (blockIdx.z == 0) ? W0 : (blockIdx.z == 1) ? W1
                   : (blockIdx.z == 2) ? W2 : W3;
    __bf16* O = (blockIdx.z == 0) ? O0 : (blockIdx.z == 1) ? O1
              : (blockIdx.z == 2) ? O2 : O3;
    const int kt = blockIdx.x * 32, nt = blockIdx.y * 32;
    const int tx = threadIdx.x & 31, ty = threadIdx.x >> 5;   // 32 x 8
#pragma unroll
    for (int i = 0; i < 4; ++i) {
        const int r = ty + i * 8;
        tile[r][tx] = (__bf16)W[(size_t)(kt + r) * D_ + nt + tx];
    }
    __syncthreads();
#pragma unroll
    for (int i = 0; i < 4; ++i) {
        const int r = ty + i * 8;
        O[(size_t)(nt + r) * D_ + kt + tx] = tile[tx][r];
    }
}

// ---------------------------------------------------------------------------
// Kernel 1: QKV projection (tiled WMMA GEMM).
// Q,K stored [B,H,T,HD]; V stored transposed [B,H,HD,T] for attention.
// Uniform `which` branch hoisted out of the store loops (SALU/exec hygiene).
// ---------------------------------------------------------------------------
__global__ __launch_bounds__(256)
void qkv_gemm_kernel(const __bf16* __restrict__ xb,
                     const __bf16* __restrict__ Wqt, const __bf16* __restrict__ Wkt,
                     const __bf16* __restrict__ Wvt,
                     __bf16* __restrict__ Q, __bf16* __restrict__ K,
                     __bf16* __restrict__ Vt) {
    __shared__ __bf16 As[2][BM * LDT];
    __shared__ __bf16 Bs[2][BM * LDT];

    const int which = blockIdx.z;
    const __bf16* Wt = (which == 0) ? Wqt : (which == 1) ? Wkt : Wvt;

    const int row0 = blockIdx.x * BM;
    const int col0 = blockIdx.y * BN;
    v8f acc[4][2];
    gemm_mainloop(xb, Wt, row0, col0, D_, D_, D_, &As[0][0], &Bs[0][0], acc);

    const int wave = threadIdx.x >> 5, lane = threadIdx.x & 31;
    const int wm = wave >> 2, wn = wave & 3;
    const int n_ = lane & 15, half = lane >> 4;

    if (which != 2) {
        // Q or K: [B,H,T,HD]
        __bf16* O = (which == 0) ? Q : K;
#pragma unroll
        for (int mt = 0; mt < 4; ++mt)
#pragma unroll
            for (int nt = 0; nt < 2; ++nt)
#pragma unroll
                for (int r = 0; r < 8; ++r) {
                    const int grow = row0 + wm * 64 + mt * 16 + r + 8 * half;
                    const int gcol = col0 + wn * 32 + nt * 16 + n_;
                    const int b_ = grow >> 11, t = grow & 2047;
                    const int h  = gcol >> 6,  hd = gcol & 63;
                    O[(((size_t)(b_ * H_ + h)) * T_ + t) * HD_ + hd] =
                        (__bf16)acc[mt][nt][r];
                }
    } else {
        // V transposed: [B,H,HD,T]
#pragma unroll
        for (int mt = 0; mt < 4; ++mt)
#pragma unroll
            for (int nt = 0; nt < 2; ++nt)
#pragma unroll
                for (int r = 0; r < 8; ++r) {
                    const int grow = row0 + wm * 64 + mt * 16 + r + 8 * half;
                    const int gcol = col0 + wn * 32 + nt * 16 + n_;
                    const int b_ = grow >> 11, t = grow & 2047;
                    const int h  = gcol >> 6,  hd = gcol & 63;
                    Vt[(((size_t)(b_ * H_ + h)) * HD_ + hd) * T_ + t] =
                        (__bf16)acc[mt][nt][r];
                }
    }
}

// ---------------------------------------------------------------------------
// Kernel 2: causal flash attention. One wave per (b*h, 32-query block):
// two 16-row query sub-tiles share every K/V fragment (2x reuse).
// Per 32-key chunk: 8 S-WMMAs + 8 PV-WMMAs against 8 fragment loads.
// ---------------------------------------------------------------------------
__global__ __launch_bounds__(32)
void flash_attn_kernel(const __bf16* __restrict__ Q, const __bf16* __restrict__ K,
                       const __bf16* __restrict__ Vt, __bf16* __restrict__ Ctx) {
    __shared__ __bf16 plds[2][16 * 32];

    const int qb = blockIdx.x;           // 32-query block, 0..T/32-1
    const int bh = blockIdx.y;           // 0..31
    const int q0 = qb * 32;
    const __bf16* Qp = Q  + (size_t)bh * T_ * HD_;
    const __bf16* Kp = K  + (size_t)bh * T_ * HD_;
    const __bf16* Vp = Vt + (size_t)bh * HD_ * T_;   // [HD][T]

    const int lane = threadIdx.x & 31;
    const int n    = lane & 15;
    const int half = lane >> 4;

    v16bf qa[2][2];
#pragma unroll
    for (int qt = 0; qt < 2; ++qt) {
        qa[qt][0] = load_A_frag(Qp + (size_t)(q0 + qt * 16) * HD_ + 0,  HD_);
        qa[qt][1] = load_A_frag(Qp + (size_t)(q0 + qt * 16) * HD_ + 32, HD_);
    }

    v8f o[2][4];
    float mrow[2][8], lrow[2][8];
#pragma unroll
    for (int qt = 0; qt < 2; ++qt)
#pragma unroll
        for (int r = 0; r < 8; ++r) {
            mrow[qt][r] = -__builtin_inff();
            lrow[qt][r] = 0.0f;
            o[qt][r & 3] = (v8f){};      // zero all 4 (r&3 cycles 0..3 twice)
        }

    const float scale   = 0.125f;        // 1/sqrt(HD)
    const int   nchunks = qb + 1;        // keys 0 .. q0+31

    for (int ch = 0; ch < nchunks; ++ch) {
        const int k0 = ch * 32;

        // shared K/V fragments for this chunk
        const v16bf kb0a = load_Bt_frag(Kp + (size_t)k0 * HD_,             HD_);
        const v16bf kb0b = load_Bt_frag(Kp + (size_t)k0 * HD_ + 32,        HD_);
        const v16bf kb1a = load_Bt_frag(Kp + (size_t)(k0 + 16) * HD_,      HD_);
        const v16bf kb1b = load_Bt_frag(Kp + (size_t)(k0 + 16) * HD_ + 32, HD_);
        v16bf vb[4];
#pragma unroll
        for (int c = 0; c < 4; ++c)
            vb[c] = load_Bt_frag(Vp + (size_t)(c * 16) * T_ + k0, T_);

#pragma unroll
        for (int qt = 0; qt < 2; ++qt) {
            const int qbase = q0 + qt * 16;

            v8f s0 = {}, s1 = {};
            s0 = wmma_bf16(qa[qt][0], kb0a, s0);
            s0 = wmma_bf16(qa[qt][1], kb0b, s0);
            s1 = wmma_bf16(qa[qt][0], kb1a, s1);
            s1 = wmma_bf16(qa[qt][1], kb1b, s1);

            const bool need_mask = (k0 + 31 > qbase);
#pragma unroll
            for (int r = 0; r < 8; ++r) {
                s0[r] *= scale;
                s1[r] *= scale;
                if (need_mask) {
                    const int qrow = qbase + r + 8 * half;
                    if (k0 + n      > qrow) s0[r] = -__builtin_inff();
                    if (k0 + 16 + n > qrow) s1[r] = -__builtin_inff();
                }
            }

#pragma unroll
            for (int r = 0; r < 8; ++r) {
                float tmax = fmaxf(s0[r], s1[r]);
#pragma unroll
                for (int off = 8; off >= 1; off >>= 1)
                    tmax = fmaxf(tmax, __shfl_xor(tmax, off, 16));
                const float mnew = fmaxf(mrow[qt][r], tmax);
                const float alpha = __expf(mrow[qt][r] - mnew);
                mrow[qt][r] = mnew;
                s0[r] = __expf(s0[r] - mnew);
                s1[r] = __expf(s1[r] - mnew);
                float ps = s0[r] + s1[r];
#pragma unroll
                for (int off = 8; off >= 1; off >>= 1)
                    ps += __shfl_xor(ps, off, 16);
                lrow[qt][r] = lrow[qt][r] * alpha + ps;
                o[qt][0][r] *= alpha; o[qt][1][r] *= alpha;
                o[qt][2][r] *= alpha; o[qt][3][r] *= alpha;
            }

            // P: C-layout -> A-layout staging
#pragma unroll
            for (int r = 0; r < 8; ++r) {
                const int row = r + 8 * half;
                plds[qt][row * 32 + n]      = (__bf16)s0[r];
                plds[qt][row * 32 + 16 + n] = (__bf16)s1[r];
            }
        }

        __syncthreads();
#pragma unroll
        for (int qt = 0; qt < 2; ++qt) {
            const v16bf pa = load_A_frag(plds[qt], 32);
#pragma unroll
            for (int c = 0; c < 4; ++c)
                o[qt][c] = wmma_bf16(pa, vb[c], o[qt][c]);
        }
        __syncthreads();
    }

    const int b_ = bh >> 4;
    const int h  = bh & 15;
#pragma unroll
    for (int qt = 0; qt < 2; ++qt)
#pragma unroll
        for (int r = 0; r < 8; ++r) {
            const float inv = 1.0f / lrow[qt][r];
            const int   t   = q0 + qt * 16 + r + 8 * half;
            __bf16* dst = Ctx + ((size_t)b_ * T_ + t) * D_ + h * HD_;
            dst[n]      = (__bf16)(o[qt][0][r] * inv);
            dst[16 + n] = (__bf16)(o[qt][1][r] * inv);
            dst[32 + n] = (__bf16)(o[qt][2][r] * inv);
            dst[48 + n] = (__bf16)(o[qt][3][r] * inv);
        }
}

// ---------------------------------------------------------------------------
// Kernel 3: output projection  out = ctx @ Wo + bo   (fp32 output)
// ---------------------------------------------------------------------------
__global__ __launch_bounds__(256)
void out_proj_kernel(const __bf16* __restrict__ Ctx, const __bf16* __restrict__ Wot,
                     const float* __restrict__ bo, float* __restrict__ out) {
    __shared__ __bf16 As[2][BM * LDT];
    __shared__ __bf16 Bs[2][BM * LDT];

    const int row0 = blockIdx.x * BM;
    const int col0 = blockIdx.y * BN;
    v8f acc[4][2];
    gemm_mainloop(Ctx, Wot, row0, col0, D_, D_, D_, &As[0][0], &Bs[0][0], acc);

    const int wave = threadIdx.x >> 5, lane = threadIdx.x & 31;
    const int wm = wave >> 2, wn = wave & 3;
    const int n_ = lane & 15, half = lane >> 4;
#pragma unroll
    for (int mt = 0; mt < 4; ++mt)
#pragma unroll
        for (int nt = 0; nt < 2; ++nt) {
            const int gcol = col0 + wn * 32 + nt * 16 + n_;
            const float bias = bo[gcol];
#pragma unroll
            for (int r = 0; r < 8; ++r) {
                const int grow = row0 + wm * 64 + mt * 16 + r + 8 * half;
                out[(size_t)grow * D_ + gcol] = acc[mt][nt][r] + bias;
            }
        }
}

// ---------------------------------------------------------------------------
extern "C" void kernel_launch(void* const* d_in, const int* in_sizes, int n_in,
                              void* d_out, int out_size, void* d_ws, size_t ws_size,
                              hipStream_t stream) {
    const float* x  = (const float*)d_in[0];
    const float* Wq = (const float*)d_in[1];
    const float* Wk = (const float*)d_in[2];
    const float* Wv = (const float*)d_in[3];
    const float* Wo = (const float*)d_in[4];
    const float* bo = (const float*)d_in[5];
    float* out = (float*)d_out;

    const size_t NE = (size_t)BT_ * D_;          // 4M elements
    __bf16* xb  = (__bf16*)d_ws;
    __bf16* Wqt = xb  + NE;
    __bf16* Wkt = Wqt + NE;
    __bf16* Wvt = Wkt + NE;
    __bf16* Wot = Wvt + NE;
    __bf16* Qb  = Wot + NE;
    __bf16* Kb  = Qb  + NE;
    __bf16* Vtb = Kb  + NE;
    __bf16* Cb  = Vtb + NE;

    cast_x_kernel<<<dim3((unsigned)(NE / (256 * 8))), dim3(256), 0, stream>>>(x, xb);
    transpose_cast_kernel<<<dim3(D_ / 32, D_ / 32, 4), dim3(256), 0, stream>>>(
        Wq, Wk, Wv, Wo, Wqt, Wkt, Wvt, Wot);
    qkv_gemm_kernel<<<dim3(BT_ / BM, D_ / BN, 3), dim3(256), 0, stream>>>(
        xb, Wqt, Wkt, Wvt, Qb, Kb, Vtb);
    flash_attn_kernel<<<dim3(T_ / 32, B_ * H_), dim3(32), 0, stream>>>(Qb, Kb, Vtb, Cb);
    out_proj_kernel<<<dim3(BT_ / BM, D_ / BN), dim3(256), 0, stream>>>(Cb, Wot, bo, out);
}
